// TrackHSPA_76879914598933
// MI455X (gfx1250) — compile-verified, matrix operand order checked by reading, not run on hardware
//
#include <hip/hip_runtime.h>
#include <hip/hip_bf16.h>
#include <stdint.h>

// ---------------------------------------------------------------------------
// Types for WMMA fragments
// ---------------------------------------------------------------------------
typedef __attribute__((ext_vector_type(16))) _Float16 v16h;
typedef __attribute__((ext_vector_type(8)))  _Float16 v8h;
typedef __attribute__((ext_vector_type(8)))  float    v8f;

#define BATCH 2
#define CIN   256
#define CR    64
#define HH    64
#define WW    64
#define NPIX  4096  // H*W

// ---------------------------------------------------------------------------
// Fragment loaders. Layouts per CDNA5 ISA 7.12.2 (wave32):
//  A (16x32 f16): lane l -> row M = l&15; halves 0..7 = K = k0 + 8*(l>>4) + i,
//                 halves 8..15 = K + 16.
//  B (32x16 f16): symmetric (lane -> col N = l&15, same K striping), so a
//                 K-contiguous [col][k] buffer serves B directly.
//  C/D (16x16 f32): lane l -> col = l&15; vgpr r -> row = r + 8*(l>>4).
// Source buffer must be row-major with K contiguous; two b128 loads per frag.
// ---------------------------------------------------------------------------
__device__ inline v16h frag_ld(const _Float16* mat, int pitch, int row0, int k0) {
  int lane = threadIdx.x & 31;
  int m  = lane & 15;
  int hi = lane >> 4;
  const _Float16* p = mat + (size_t)(row0 + m) * pitch + k0 + hi * 8;
  v8h lo = *(const v8h*)p;
  v8h hh = *(const v8h*)(p + 16);
  v16h r;
#pragma unroll
  for (int i = 0; i < 8; ++i) { r[i] = lo[i]; r[i + 8] = hh[i]; }
  return r;
}

__device__ inline v8f wmma_f16(v16h a, v16h b, v8f c) {
  return __builtin_amdgcn_wmma_f32_16x16x32_f16(false, a, false, b,
                                                (short)0, c, false, false);
}

// ---------------------------------------------------------------------------
// 0) Repack conv weights: conv_w[cr][c][ky][kx] f32 -> wpack[tap][cr][c] f16
// ---------------------------------------------------------------------------
__global__ void pack_w_kernel(const float* __restrict__ conv_w,
                              _Float16* __restrict__ wpack) {
  int idx = blockIdx.x * blockDim.x + threadIdx.x;   // (tap*64+cr)*256 + c
  if (idx >= 9 * CR * CIN) return;
  int tap = idx >> 14;
  int cr  = (idx >> 8) & 63;
  int c   = idx & 255;
  wpack[idx] = (_Float16)conv_w[((size_t)cr * CIN + c) * 9 + tap];
}

// ---------------------------------------------------------------------------
// 0b) x f32 -> f16 ([b][c][n], n contiguous; feeds conv staging + out-GEMM A)
// ---------------------------------------------------------------------------
__global__ void cvt_x_kernel(const float* __restrict__ x,
                             _Float16* __restrict__ xh, int n) {
  int i = blockIdx.x * blockDim.x + threadIdx.x;
  if (i < n) xh[i] = (_Float16)x[i];
}

// ---------------------------------------------------------------------------
// 1) Fused conv3x3(pad1) + bias + BN + PReLU as shifted GEMMs.
//    One workgroup = one (batch, image row).  The dx shift is folded into the
//    LDS read address: one padded row-slice [pixel -1..64][c] is staged per
//    dy and serves all three dx taps (3 stagings instead of 9; rows shift by
//    whole 528B rows so b128 alignment is preserved).
//    A wave's two tiles share tn -> one B fragment feeds two WMMAs.
//    Writes eT[b][n][d] f16 (d contiguous) -- serves A and B of sim GEMM.
// ---------------------------------------------------------------------------
#define BP 264  // LDS pitch in halves (528B = 33*16B: rows b128-aligned)

__global__ __launch_bounds__(256) void conv_kernel(
    const _Float16* __restrict__ xh, const float* __restrict__ conv_b,
    const float* __restrict__ gamma, const float* __restrict__ beta,
    const float* __restrict__ mean, const float* __restrict__ var,
    const float* __restrict__ prelu, const _Float16* __restrict__ wpack,
    _Float16* __restrict__ eT) {
  __shared__ _Float16 Blds[66 * BP];   // [pixel+1][channel], ~34.8 KB

  int blk = blockIdx.x;          // 0..127
  int b = blk >> 6, h = blk & 63;
  int tid  = threadIdx.x;
  int wave = tid >> 5, lane = tid & 31;

  // 16 tiles (4x4) of 16x16 over the 64x64 output; wave owns {wave, wave+8}
  int tn  = wave & 3;            // shared column tile
  int tm0 = wave >> 2;           // row tiles tm0 and tm0+2
  v8f acc0 = {}, acc1 = {};

  for (int dy = 0; dy < 3; ++dy) {
    int hs = h + dy - 1;
    if (hs < 0 || hs >= HH) continue;          // uniform across workgroup

    __syncthreads();
    // stage padded row slice: LDS[pix][c] = xh[b,c,hs,pix-1], zeros at ends
    for (int idx = tid; idx < CIN * 66; idx += 256) {
      int c = idx / 66, pix = idx - c * 66;    // pix fast => coalesced gmem
      _Float16 v = (_Float16)0.f;
      if (pix >= 1 && pix <= WW)
        v = xh[((size_t)(b * CIN + c)) * NPIX + hs * WW + pix - 1];
      Blds[pix * BP + c] = v;
    }
    __syncthreads();

#pragma unroll
    for (int dx = 0; dx < 3; ++dx) {
      const _Float16* wt = wpack + (size_t)(dy * 3 + dx) * CR * CIN;
#pragma unroll
      for (int k0 = 0; k0 < CIN; k0 += 32) {
        v16h a0 = frag_ld(wt, CIN, tm0 * 16, k0);        // global (L2-hot)
        v16h a1 = frag_ld(wt, CIN, (tm0 + 2) * 16, k0);
        v16h bf = frag_ld(Blds, BP, tn * 16 + dx, k0);   // ds, shared
        acc0 = wmma_f16(a0, bf, acc0);
        acc1 = wmma_f16(a1, bf, acc1);
      }
    }
  }

  // epilogue: bias + BN + PReLU, store transposed eT[n][cr]
  int hi = lane >> 4, cl = lane & 15;
  int n = h * WW + tn * 16 + cl;
#pragma unroll
  for (int t = 0; t < 2; ++t) {
    int tm = tm0 + 2 * t;
    const v8f& a = t ? acc1 : acc0;
#pragma unroll
    for (int r = 0; r < 8; ++r) {
      int cr = tm * 16 + r + 8 * hi;
      float v = a[r] + conv_b[cr];
      float sc = gamma[cr] * rsqrtf(var[cr] + 1e-5f);
      v = (v - mean[cr]) * sc + beta[cr];
      v = (v >= 0.f) ? v : prelu[0] * v;
      eT[((size_t)b * NPIX + n) * CR + cr] = (_Float16)v;
    }
  }
}

// ---------------------------------------------------------------------------
// 2) sim = e^T e : M=N=4096, K=64. One wave -> 64x64 block (16 tiles).
//    eT is symmetric source for A and B fragments; no LDS (512KB, L2-hot).
// ---------------------------------------------------------------------------
__global__ __launch_bounds__(256) void sim_kernel(const _Float16* __restrict__ eTb,
                                                  float* __restrict__ sim) {
  int wid  = blockIdx.x * 8 + (threadIdx.x >> 5);  // 0..4095
  int lane = threadIdx.x & 31;
  int bm = wid >> 6, bn = wid & 63;
  v8f acc[4][4] = {};
#pragma unroll
  for (int k0 = 0; k0 < CR; k0 += 32) {
    v16h A[4], B[4];
#pragma unroll
    for (int i = 0; i < 4; ++i) A[i] = frag_ld(eTb, CR, bm * 64 + i * 16, k0);
#pragma unroll
    for (int j = 0; j < 4; ++j) B[j] = frag_ld(eTb, CR, bn * 64 + j * 16, k0);
#pragma unroll
    for (int i = 0; i < 4; ++i)
#pragma unroll
      for (int j = 0; j < 4; ++j) acc[i][j] = wmma_f16(A[i], B[j], acc[i][j]);
  }
  int hi = lane >> 4, cl = lane & 15;
#pragma unroll
  for (int i = 0; i < 4; ++i)
#pragma unroll
    for (int j = 0; j < 4; ++j) {
      int col = bn * 64 + j * 16 + cl;
#pragma unroll
      for (int r = 0; r < 8; ++r) {
        int row = bm * 64 + i * 16 + r + 8 * hi;
        sim[(size_t)row * NPIX + col] = acc[i][j][r];
      }
    }
}

// ---------------------------------------------------------------------------
// 3) sparsemax threshold tau per row via bisection (sparsemax is shift-
//    invariant, so the reference's row-max subtraction folds away; tau* lies
//    in [zmax-1, zmax]).  One workgroup per row.
// ---------------------------------------------------------------------------
__global__ __launch_bounds__(256) void tau_kernel(const float* __restrict__ sim,
                                                  float* __restrict__ tau) {
  __shared__ float red[256];
  int row = blockIdx.x, tid = threadIdx.x;
  const float* z = sim + (size_t)row * NPIX;
  float v[16], mx = -3.0e38f;
#pragma unroll
  for (int i = 0; i < 16; ++i) { v[i] = z[tid + i * 256]; mx = fmaxf(mx, v[i]); }
  red[tid] = mx; __syncthreads();
  for (int s = 128; s > 0; s >>= 1) {
    if (tid < s) red[tid] = fmaxf(red[tid], red[tid + s]);
    __syncthreads();
  }
  float zmax = red[0]; __syncthreads();
  float lo = zmax - 1.0f, hi = zmax;
  for (int it = 0; it < 30; ++it) {
    float mid = 0.5f * (lo + hi);
    float s = 0.f;
#pragma unroll
    for (int i = 0; i < 16; ++i) s += fmaxf(v[i] - mid, 0.f);
    red[tid] = s; __syncthreads();
    for (int st = 128; st > 0; st >>= 1) {
      if (tid < st) red[tid] += red[tid + st];
      __syncthreads();
    }
    float tot = red[0]; __syncthreads();
    if (tot > 1.0f) lo = mid; else hi = mid;   // sum is decreasing in tau
  }
  if (tid == 0) tau[row] = 0.5f * (lo + hi);
}

// ---------------------------------------------------------------------------
// 4) p = max(sim - tau, 0) -> f16, transposed to pT[m][n] (n contiguous) so
//    the out-GEMM's B fragments are contiguous b128 loads. 64x64 LDS tiles,
//    coalesced on both global sides.
// ---------------------------------------------------------------------------
__global__ __launch_bounds__(256) void psparse_kernel(const float* __restrict__ sim,
                                                      const float* __restrict__ tau,
                                                      _Float16* __restrict__ pT) {
  __shared__ _Float16 T[64 * 80];        // rows 160B => 16B-aligned
  int tile = blockIdx.x;                 // 64x64 grid of tiles
  int n0 = (tile >> 6) * 64;             // row block (n)
  int m0 = (tile & 63) * 64;             // col block (m)
  int tid = threadIdx.x;
  for (int idx = tid; idx < 4096; idx += 256) {
    int nl = idx >> 6, ml = idx & 63;    // ml fast => coalesced sim read
    float z = sim[(size_t)(n0 + nl) * NPIX + m0 + ml];
    float p = fmaxf(z - tau[n0 + nl], 0.f);
    T[ml * 80 + nl] = (_Float16)p;
  }
  __syncthreads();
  for (int idx = tid; idx < 4096; idx += 256) {
    int ml = idx >> 6, nl = idx & 63;    // nl fast => coalesced pT write
    pT[(size_t)(m0 + ml) * NPIX + n0 + nl] = T[ml * 80 + nl];
  }
}

// ---------------------------------------------------------------------------
// 5) out[c][m] = sum_n x[c][n] * p[n][m]  (M=256, N=4096, K=4096) + residual.
//    One wave -> 64x64 block.  Register ping-pong (unroll-by-2): fragments
//    for k+32 are fetched while the 16 WMMAs for k execute, so the wave never
//    waits on a just-issued load.
// ---------------------------------------------------------------------------
__global__ __launch_bounds__(256) void outgemm_kernel(
    const _Float16* __restrict__ xh_b, const _Float16* __restrict__ pT,
    const float* __restrict__ x_b, float* __restrict__ out_b) {
  int wid  = blockIdx.x * 8 + (threadIdx.x >> 5);  // 0..255
  int lane = threadIdx.x & 31;
  int bm = wid >> 6;                   // 0..3 over C=256
  int bn = wid & 63;                   // 0..63 over 4096
  v8f acc[4][4] = {};
  v16h A0[4], B0[4], A1[4], B1[4];

#pragma unroll
  for (int i = 0; i < 4; ++i) A0[i] = frag_ld(xh_b, NPIX, bm * 64 + i * 16, 0);
#pragma unroll
  for (int j = 0; j < 4; ++j) B0[j] = frag_ld(pT, NPIX, bn * 64 + j * 16, 0);

  for (int k0 = 0; k0 < NPIX; k0 += 64) {
    // prefetch k0+32 while computing on k0
#pragma unroll
    for (int i = 0; i < 4; ++i) A1[i] = frag_ld(xh_b, NPIX, bm * 64 + i * 16, k0 + 32);
#pragma unroll
    for (int j = 0; j < 4; ++j) B1[j] = frag_ld(pT, NPIX, bn * 64 + j * 16, k0 + 32);
#pragma unroll
    for (int i = 0; i < 4; ++i)
#pragma unroll
      for (int j = 0; j < 4; ++j) acc[i][j] = wmma_f16(A0[i], B0[j], acc[i][j]);

    if (k0 + 64 < NPIX) {
#pragma unroll
      for (int i = 0; i < 4; ++i) A0[i] = frag_ld(xh_b, NPIX, bm * 64 + i * 16, k0 + 64);
#pragma unroll
      for (int j = 0; j < 4; ++j) B0[j] = frag_ld(pT, NPIX, bn * 64 + j * 16, k0 + 64);
    }
#pragma unroll
    for (int i = 0; i < 4; ++i)
#pragma unroll
      for (int j = 0; j < 4; ++j) acc[i][j] = wmma_f16(A1[i], B1[j], acc[i][j]);
  }

  int hi = lane >> 4, cl = lane & 15;
#pragma unroll
  for (int i = 0; i < 4; ++i)
#pragma unroll
    for (int j = 0; j < 4; ++j) {
      int col = bn * 64 + j * 16 + cl;
#pragma unroll
      for (int r = 0; r < 8; ++r) {
        int c = bm * 64 + i * 16 + r + 8 * hi;
        size_t o = (size_t)c * NPIX + col;
        out_b[o] = acc[i][j][r] + x_b[o];   // residual add
      }
    }
}

// ---------------------------------------------------------------------------
// Host-side launch.  Workspace layout (256B aligned), ~101.5 MB total:
//   wpack  9*64*256      f16   288 KB
//   xh     2*256*4096    f16     4 MB
//   eT     2*4096*64     f16     1 MB
//   tau    4096          f32    16 KB
//   sim    4096*4096     f32    64 MB   (reused per batch)
//   pT     4096*4096     f16    32 MB   (reused per batch)
// ---------------------------------------------------------------------------
static inline size_t align256(size_t v) { return (v + 255) & ~(size_t)255; }

extern "C" void kernel_launch(void* const* d_in, const int* in_sizes, int n_in,
                              void* d_out, int out_size, void* d_ws, size_t ws_size,
                              hipStream_t stream) {
  const float* x      = (const float*)d_in[0];
  const float* conv_w = (const float*)d_in[1];
  const float* conv_b = (const float*)d_in[2];
  const float* gamma  = (const float*)d_in[3];
  const float* beta   = (const float*)d_in[4];
  const float* mean   = (const float*)d_in[5];
  const float* var    = (const float*)d_in[6];
  const float* prelu  = (const float*)d_in[7];
  float* out = (float*)d_out;

  char* ws = (char*)d_ws;
  size_t off = 0;
  _Float16* wpack = (_Float16*)(ws + off); off = align256(off + (size_t)9 * CR * CIN * 2);
  _Float16* xh    = (_Float16*)(ws + off); off = align256(off + (size_t)BATCH * CIN * NPIX * 2);
  _Float16* eT    = (_Float16*)(ws + off); off = align256(off + (size_t)BATCH * NPIX * CR * 2);
  float*    tau   = (float*)   (ws + off); off = align256(off + (size_t)NPIX * 4);
  float*    sim   = (float*)   (ws + off); off = align256(off + (size_t)NPIX * NPIX * 4);
  _Float16* pT    = (_Float16*)(ws + off); off = align256(off + (size_t)NPIX * NPIX * 2);
  (void)ws_size; (void)n_in; (void)in_sizes; (void)out_size;

  // 0) weight repack + x->f16 (cvt_x must precede conv: stream-ordered)
  pack_w_kernel<<<(9 * CR * CIN + 255) / 256, 256, 0, stream>>>(conv_w, wpack);
  {
    int n = BATCH * CIN * NPIX;
    cvt_x_kernel<<<(n + 255) / 256, 256, 0, stream>>>(x, xh, n);
  }

  // 1) conv + BN + PReLU -> eT (both batches)
  conv_kernel<<<BATCH * HH, 256, 0, stream>>>(xh, conv_b, gamma, beta, mean, var,
                                              prelu, wpack, eT);

  // per-batch pipeline (sim/pT buffers reused; stream-ordered)
  for (int b = 0; b < BATCH; ++b) {
    const _Float16* eTb = eT + (size_t)b * NPIX * CR;
    const _Float16* xhb = xh + (size_t)b * CIN * NPIX;
    const float*    xb  = x  + (size_t)b * CIN * NPIX;
    float*          ob  = out + (size_t)b * CIN * NPIX;

    sim_kernel    <<<(64 * 64) / 8, 256, 0, stream>>>(eTb, sim);
    tau_kernel    <<<NPIX,          256, 0, stream>>>(sim, tau);
    psparse_kernel<<<64 * 64,       256, 0, stream>>>(sim, tau, pT);
    outgemm_kernel<<<(4 * 64) / 8,  256, 0, stream>>>(xhb, pT, xb, ob);
  }
}
